// SelfAttention_33646773796936
// MI455X (gfx1250) — compile-verified, matrix operand order checked by reading, not run on hardware
//
#include <hip/hip_runtime.h>

#define N_PTS 8192
#define BATCH 2
#define KNN 16
#define EPS 1e-5f
#define SLOPE 0.2f

typedef __attribute__((ext_vector_type(16))) _Float16 v16h;
typedef __attribute__((ext_vector_type(8)))  float    v8f;
typedef __attribute__((ext_vector_type(4)))  unsigned int u32x4;

// Load a 16-element f16 WMMA A/B fragment from a row-major (stride CIN) slab.
// Per ISA 7.12.2 (16-bit A 16x32 / B 32x16): lane = M (A) or N (B); the
// half-wave selects K-half (khalf = 0 or 8 halves); fragment = two contiguous
// 16-byte runs: halves [khalf, khalf+8) and [16+khalf, 16+khalf+8).
// Both runs are 16B-aligned -> exactly two b128 loads.
__device__ __forceinline__ v16h load_frag(const _Float16* rowbase, int khalf) {
  union { u32x4 q[2]; v16h v; } u;
  u.q[0] = *reinterpret_cast<const u32x4*>(rowbase + khalf);
  u.q[1] = *reinterpret_cast<const u32x4*>(rowbase + 16 + khalf);
  return u.v;
}

// ---------------------------------------------------------------- KNN -------
__global__ void __launch_bounds__(256) knn_kernel(const float* __restrict__ coords,
                                                  int* __restrict__ nnidx) {
  __shared__ float sx[256], sy[256], sz[256];
  const int tid = threadIdx.x;
  const int b   = blockIdx.y;
  const int n   = blockIdx.x * 256 + tid;
  const float* cb = coords + (size_t)b * 3 * N_PTS;
  const float px = cb[n], py = cb[N_PTS + n], pz = cb[2 * N_PTS + n];

  float bd[17]; int bi[17];
#pragma unroll
  for (int t = 0; t < 17; ++t) { bd[t] = 3.4e38f; bi[t] = 0; }

  for (int tile = 0; tile < N_PTS; tile += 256) {
    __syncthreads();
    sx[tid] = cb[tile + tid];
    sy[tid] = cb[N_PTS + tile + tid];
    sz[tid] = cb[2 * N_PTS + tile + tid];
    __syncthreads();
    for (int jj = 0; jj < 256; ++jj) {
      const float dx = px - sx[jj], dy = py - sy[jj], dz = pz - sz[jj];
      const float d = dx * dx + dy * dy + dz * dz;
      if (d < bd[16]) {
        bd[16] = d; bi[16] = tile + jj;
#pragma unroll
        for (int t = 16; t > 0; --t) {
          if (bd[t] < bd[t - 1]) {
            float td = bd[t]; bd[t] = bd[t-1]; bd[t-1] = td;
            int   ti = bi[t]; bi[t] = bi[t-1]; bi[t-1] = ti;
          }
        }
      }
    }
  }
#pragma unroll
  for (int k = 0; k < KNN; ++k)
    nnidx[((size_t)b * N_PTS + n) * KNN + k] = bi[k + 1];  // drop self
}

// --------------------------------------------- fused edge-gather + WMMA -----
// Block: 128 threads = 4 waves; 4 points/block; columns = point*16 + k.
// Dynamic LDS: [edge tile COLS*CIN | weight slab COUT*CIN], both f16.
// B fragments live in VGPRs; A fragments stream from the LDS weight slab
// through a 2-deep software pipeline (load kt+1, wmma kt).
template <int COUT>
__global__ void __launch_bounds__(128) edgeconv_kernel(
    const float* __restrict__ fin,      // (B, 128, N)
    const int*   __restrict__ nnidx,    // (B, N, K)
    const _Float16* __restrict__ wh,    // (COUT, 256) f16
    float* __restrict__ maxout,         // (B, COUT, N)  pre-norm max_k
    float* __restrict__ sum, float* __restrict__ sq) {
  constexpr int CIN = 256, COLS = 64, NKT = CIN / 32;
  extern __shared__ _Float16 smem[];
  _Float16* edge = smem;                // [col][cin], 32 KB
  _Float16* wlds = smem + COLS * CIN;   // [row][cin], COUT*CIN*2 bytes

  const int tid = threadIdx.x;
  const int b = blockIdx.y;
  const int pbase = blockIdx.x * 4;

  // warm L2 for the weight slab while the scattered gather runs
  __builtin_prefetch(wh + (size_t)tid * 256, 0, 1);

  // stage W -> LDS (coalesced b128 copies)
  {
    const u32x4* src = reinterpret_cast<const u32x4*>(wh);
    u32x4* dst = reinterpret_cast<u32x4*>(wlds);
#pragma unroll 4
    for (int i = tid; i < COUT * CIN / 8; i += 128) dst[i] = src[i];
  }

  // gather: 8 consecutive channels of one column -> one ds_store_b128
  // (range boundary 128 is a multiple of 8, so a chunk never straddles)
  for (int e = tid; e < COLS * (CIN / 8); e += 128) {
    const int col = e >> 5;               // CIN/8 == 32 chunks per column
    const int cin0 = (e & 31) * 8;
    const int p = col >> 4, k = col & 15;
    const int n = pbase + p;
    union { u32x4 q; _Float16 h[8]; } pk;
    if (cin0 < 128) {
      const float* fc = fin + ((size_t)b * 128 + cin0) * N_PTS;
#pragma unroll
      for (int j = 0; j < 8; ++j) pk.h[j] = (_Float16)fc[(size_t)j * N_PTS + n];
    } else {
      const float* fc = fin + ((size_t)b * 128 + (cin0 - 128)) * N_PTS;
      const int nb = nnidx[((size_t)b * N_PTS + n) * KNN + k];
#pragma unroll
      for (int j = 0; j < 8; ++j)
        pk.h[j] = (_Float16)(fc[(size_t)j * N_PTS + nb] - fc[(size_t)j * N_PTS + n]);
    }
    *reinterpret_cast<u32x4*>(&edge[col * CIN + cin0]) = pk.q;
  }
  __syncthreads();

  const int wave = tid >> 5, lane = tid & 31;
  const int hrow = (lane >= 16) ? 8 : 0;      // K-half for frags; +8 row in C/D
  const int point = pbase + wave;
  const _Float16* bbase = &edge[(wave * 16 + (lane & 15)) * CIN];

  // B fragments resident in VGPRs (8 x 8 = 64 VGPRs)
  v16h bfr[NKT];
#pragma unroll
  for (int kt = 0; kt < NKT; ++kt) bfr[kt] = load_frag(bbase + kt * 32, hrow);

#pragma unroll 1
  for (int rt = 0; rt < COUT / 16; ++rt) {
    const _Float16* abase = wlds + (size_t)(rt * 16 + (lane & 15)) * CIN;

    // 2-deep pipelined A stream from LDS
    v16h afr[2];
    afr[0] = load_frag(abase, hrow);
    v8f acc = {0.f, 0.f, 0.f, 0.f, 0.f, 0.f, 0.f, 0.f};
#pragma unroll
    for (int kt = 0; kt < NKT; ++kt) {
      if (kt + 1 < NKT) afr[(kt + 1) & 1] = load_frag(abase + (kt + 1) * 32, hrow);
      acc = __builtin_amdgcn_wmma_f32_16x16x32_f16(false, afr[kt & 1], false,
                                                   bfr[kt], (short)0, acc,
                                                   false, false);
    }

    // C/D layout: VGPR r holds row rt*16+r (+8 upper half), 16 cols in 16 lanes.
#pragma unroll
    for (int r = 0; r < 8; ++r) {
      float m = acc[r], s = acc[r], q = acc[r] * acc[r];
#pragma unroll
      for (int msk = 1; msk < 16; msk <<= 1) {
        m = fmaxf(m, __shfl_xor(m, msk, 16));
        s += __shfl_xor(s, msk, 16);
        q += __shfl_xor(q, msk, 16);
      }
      if ((lane & 15) == 0) {
        const int c = rt * 16 + r + hrow;
        maxout[((size_t)b * COUT + c) * N_PTS + point] = m;
        atomicAdd(&sum[b * COUT + c], s);
        atomicAdd(&sq[b * COUT + c], q);
      }
    }
  }
}

// --------------------------------------------- final 128x512 WMMA GEMM ------
__global__ void __launch_bounds__(128) conv3_kernel(
    const float* __restrict__ x0, const float* __restrict__ x1,
    const float* __restrict__ x2,
    const _Float16* __restrict__ wh,    // (128, 512) f16
    float* __restrict__ hout,           // (B, 128, N) pre-norm
    float* __restrict__ sum, float* __restrict__ sq) {
  constexpr int CIN = 512, COLS = 32, NKT = CIN / 32;
  extern __shared__ _Float16 smem[];
  _Float16* xt   = smem;                // [col][cin], 32 KB
  _Float16* wlds = smem + COLS * CIN;   // 128*512*2 = 128 KB

  const int tid = threadIdx.x, b = blockIdx.y;
  const int nbase = blockIdx.x * COLS;

  __builtin_prefetch(wh + (size_t)tid * 512, 0, 1);

  {
    const u32x4* src = reinterpret_cast<const u32x4*>(wh);
    u32x4* dst = reinterpret_cast<u32x4*>(wlds);
#pragma unroll 4
    for (int i = tid; i < 128 * CIN / 8; i += 128) dst[i] = src[i];
  }

  // gather concat [x0 | x1 | x2]: 8-channel chunks (boundaries 128/256 are
  // multiples of 8) -> one ds_store_b128 each
  for (int e = tid; e < COLS * (CIN / 8); e += 128) {
    const int col = e >> 6;               // CIN/8 == 64 chunks per column
    const int cin0 = (e & 63) * 8;
    const int n = nbase + col;
    const float* src;
    if (cin0 < 128)       src = x0 + ((size_t)b * 128 + cin0) * N_PTS;
    else if (cin0 < 256)  src = x1 + ((size_t)b * 128 + (cin0 - 128)) * N_PTS;
    else                  src = x2 + ((size_t)b * 256 + (cin0 - 256)) * N_PTS;
    union { u32x4 q; _Float16 h[8]; } pk;
#pragma unroll
    for (int j = 0; j < 8; ++j) pk.h[j] = (_Float16)src[(size_t)j * N_PTS + n];
    *reinterpret_cast<u32x4*>(&xt[col * CIN + cin0]) = pk.q;
  }
  __syncthreads();

  const int wave = tid >> 5, lane = tid & 31;
  const int hrow = (lane >= 16) ? 8 : 0;
  const int colBase = (wave & 1) * 16;      // waves 0,2 -> cols 0..15; 1,3 -> 16..31
  const int rtBase = (wave >> 1) * 4;       // 8 row tiles split over wave pairs
  const _Float16* bbase = &xt[(colBase + (lane & 15)) * CIN];

  // B fragments resident (16 x 8 = 128 VGPRs)
  v16h bfr[NKT];
#pragma unroll
  for (int kt = 0; kt < NKT; ++kt) bfr[kt] = load_frag(bbase + kt * 32, hrow);

#pragma unroll 1
  for (int rti = 0; rti < 4; ++rti) {
    const int rt = rtBase + rti;
    const _Float16* abase = wlds + (size_t)(rt * 16 + (lane & 15)) * CIN;

    v16h afr[2];
    afr[0] = load_frag(abase, hrow);
    v8f acc = {0.f, 0.f, 0.f, 0.f, 0.f, 0.f, 0.f, 0.f};
#pragma unroll
    for (int kt = 0; kt < NKT; ++kt) {
      if (kt + 1 < NKT) afr[(kt + 1) & 1] = load_frag(abase + (kt + 1) * 32, hrow);
      acc = __builtin_amdgcn_wmma_f32_16x16x32_f16(false, afr[kt & 1], false,
                                                   bfr[kt], (short)0, acc,
                                                   false, false);
    }

#pragma unroll
    for (int r = 0; r < 8; ++r) {
      const int row = rt * 16 + r + hrow;
      const int col = nbase + colBase + (lane & 15);
      hout[((size_t)b * 128 + row) * N_PTS + col] = acc[r];
      float s = acc[r], q = acc[r] * acc[r];
#pragma unroll
      for (int msk = 1; msk < 16; msk <<= 1) {
        s += __shfl_xor(s, msk, 16);
        q += __shfl_xor(q, msk, 16);
      }
      if ((lane & 15) == 0) {
        atomicAdd(&sum[b * 128 + row], s);
        atomicAdd(&sq[b * 128 + row], q);
      }
    }
  }
}

// ------------------------------------------------------------- helpers ------
__global__ void zero_kernel(float* p, int n) {
  int i = blockIdx.x * 256 + threadIdx.x;
  if (i < n) p[i] = 0.f;
}

__global__ void cvt_kernel(const float* __restrict__ src, _Float16* __restrict__ dst, int n) {
  int i = blockIdx.x * 256 + threadIdx.x;
  if (i < n) dst[i] = (_Float16)src[i];
}

// norm + leaky; i maps (b,c,n) -> i/N_PTS == b*COUT + c by construction
__global__ void finalize_kernel(const float* __restrict__ src, float* __restrict__ dst,
                                const float* __restrict__ sum, const float* __restrict__ sq,
                                float inv, int total) {
  int i = blockIdx.x * 256 + threadIdx.x;
  if (i >= total) return;
  const int bc = i / N_PTS;
  const float mean = sum[bc] * inv;
  const float var  = sq[bc] * inv - mean * mean;
  const float sc   = rsqrtf(var + EPS);
  const float v = (src[i] - mean) * sc;
  dst[i] = v > 0.f ? v : SLOPE * v;
}

// ------------------------------------------------------------- driver -------
extern "C" void kernel_launch(void* const* d_in, const int* in_sizes, int n_in,
                              void* d_out, int out_size, void* d_ws, size_t ws_size,
                              hipStream_t stream) {
  (void)in_sizes; (void)n_in; (void)out_size; (void)ws_size;
  const float* coords = (const float*)d_in[0];
  const float* feats  = (const float*)d_in[1];
  const float* W1 = (const float*)d_in[2];
  const float* W2 = (const float*)d_in[3];
  const float* W3 = (const float*)d_in[4];
  float* out = (float*)d_out;

  char* ws = (char*)d_ws;
  size_t off = 0;
  auto alloc = [&](size_t bytes) -> void* {
    void* p = ws + off;
    off = (off + bytes + 255) & ~(size_t)255;
    return p;
  };
  int*      idx  = (int*)alloc((size_t)BATCH * N_PTS * KNN * 4);
  _Float16* w1h  = (_Float16*)alloc(128 * 256 * 2);
  _Float16* w2h  = (_Float16*)alloc(256 * 256 * 2);
  _Float16* w3h  = (_Float16*)alloc(128 * 512 * 2);
  float*    x1   = (float*)alloc((size_t)BATCH * 128 * N_PTS * 4);
  float*    x2   = (float*)alloc((size_t)BATCH * 256 * N_PTS * 4);
  float*    h3   = (float*)alloc((size_t)BATCH * 128 * N_PTS * 4);
  float*    stats = (float*)alloc(2048 * 4);
  float* sum1 = stats;        float* sq1 = stats + 256;
  float* sum2 = stats + 512;  float* sq2 = stats + 1024;
  float* sum3 = stats + 1536; float* sq3 = stats + 1792;

  zero_kernel<<<8, 256, 0, stream>>>(stats, 2048);
  cvt_kernel<<<128, 256, 0, stream>>>(W1, w1h, 128 * 256);
  cvt_kernel<<<256, 256, 0, stream>>>(W2, w2h, 256 * 256);
  cvt_kernel<<<256, 256, 0, stream>>>(W3, w3h, 128 * 512);

  knn_kernel<<<dim3(N_PTS / 256, BATCH), 256, 0, stream>>>(coords, idx);

  const size_t sh1 = (size_t)(64 * 256 + 128 * 256) * 2;  //  96 KB
  const size_t sh2 = (size_t)(64 * 256 + 256 * 256) * 2;  // 160 KB
  const size_t sh3 = (size_t)(32 * 512 + 128 * 512) * 2;  // 160 KB

  edgeconv_kernel<128><<<dim3(N_PTS / 4, BATCH), 128, sh1, stream>>>(
      feats, idx, w1h, x1, sum1, sq1);
  finalize_kernel<<<(BATCH * 128 * N_PTS + 255) / 256, 256, 0, stream>>>(
      x1, x1, sum1, sq1, 1.f / (float)(N_PTS * KNN), BATCH * 128 * N_PTS);

  edgeconv_kernel<256><<<dim3(N_PTS / 4, BATCH), 128, sh2, stream>>>(
      x1, idx, w2h, x2, sum2, sq2);
  finalize_kernel<<<(BATCH * 256 * N_PTS + 255) / 256, 256, 0, stream>>>(
      x2, x2, sum2, sq2, 1.f / (float)(N_PTS * KNN), BATCH * 256 * N_PTS);

  conv3_kernel<<<dim3(N_PTS / 32, BATCH), 128, sh3, stream>>>(
      feats, x1, x2, w3h, h3, sum3, sq3);
  finalize_kernel<<<(BATCH * 128 * N_PTS + 255) / 256, 256, 0, stream>>>(
      h3, out, sum3, sq3, 1.f / (float)N_PTS, BATCH * 128 * N_PTS);
}
